// FullAttention_7670811591264
// MI455X (gfx1250) — matrix operand order repaired and checked by reading, not compiled, not run on hardware
//
#include <hip/hip_runtime.h>
#include <hip/hip_bf16.h>

typedef __attribute__((ext_vector_type(16))) _Float16 v16h;
typedef __attribute__((ext_vector_type(8)))  _Float16 v8h;
typedef __attribute__((ext_vector_type(4)))  _Float16 v4h;
typedef __attribute__((ext_vector_type(8)))  float    v8f;
typedef __attribute__((ext_vector_type(4)))  float    v4f;

// Problem dims fixed by the reference setup.
constexpr int B_  = 4;
constexpr int L_  = 2048;
constexpr int H_  = 16;
constexpr int E_  = 64;     // head dim for Q/K
constexpr int S_  = 2048;   // key length
constexpr int D_  = 64;     // value dim
constexpr int LT_ = L_ / 16;

constexpr int SSTR = 2056;  // score row stride (halves): dword stride 1028 ≡ 4 (mod 64)
constexpr int VSTR = 68;    // V-stage row stride (halves): 16 rows -> 544 dwords ≡ 32 (mod 64)

// ---- 16-element f16 fragment load: f32 source (convert) or f16 source (straight b128) ----
__device__ __forceinline__ v16h ld16_cvt(const float* p) {
  v4f a = *(const v4f*)(p + 0);
  v4f b = *(const v4f*)(p + 4);
  v4f c = *(const v4f*)(p + 8);
  v4f d = *(const v4f*)(p + 12);
  v16h o;
#pragma unroll
  for (int j = 0; j < 4; ++j) {
    o[j]      = (_Float16)a[j];
    o[4 + j]  = (_Float16)b[j];
    o[8 + j]  = (_Float16)c[j];
    o[12 + j] = (_Float16)d[j];
  }
  return o;
}
__device__ __forceinline__ v16h ld16_cvt(const _Float16* p) {
  v8h a = *(const v8h*)(p + 0);
  v8h b = *(const v8h*)(p + 8);
  return __builtin_shufflevector(a, b, 0, 1, 2, 3, 4, 5, 6, 7,
                                 8, 9, 10, 11, 12, 13, 14, 15);
}

// ---- one-time f32 -> f16 conversion of K/V into workspace (8 elems/thread) ----
__global__ __launch_bounds__(256)
void cvt_f32_f16(const float* __restrict__ src, _Float16* __restrict__ dst, int n8) {
  int i = blockIdx.x * 256 + threadIdx.x;
  if (i < n8) {
    v4f a = ((const v4f*)src)[2 * i];
    v4f b = ((const v4f*)src)[2 * i + 1];
    v8h o;
#pragma unroll
    for (int j = 0; j < 4; ++j) { o[j] = (_Float16)a[j]; o[4 + j] = (_Float16)b[j]; }
    ((v8h*)dst)[i] = o;
  }
}

template <typename KT>  // KT = _Float16 (workspace path) or float (fallback)
__global__ __launch_bounds__(32)
void fa_fwd_causal(const float* __restrict__ Q,
                   const KT* __restrict__ K,
                   const KT* __restrict__ V,
                   float* __restrict__ outO,   // [B,L,H,D]
                   float* __restrict__ outA) { // [B,H,L,S]
  __shared__ _Float16 Sc[16 * SSTR];   // masked scores -> unnormalized exp (f16), ~64KB
  __shared__ _Float16 Vh[32 * VSTR];   // staged V chunk (f16)
  __shared__ float    m_lds[16];       // per-row max
  __shared__ float    r_lds[16];       // per-row 1/sum

  const int lane = threadIdx.x & 31;
  const int n  = lane & 15;   // B/C/D column
  const int kh = lane >> 4;   // B-matrix K-half
  const int m  = lane & 15;   // A-matrix row
  const int ah = lane >> 4;   // A-matrix K-half
  const int mh = lane >> 4;   // C/D row-half (rows r + 8*mh)

  const int bid = blockIdx.x;
  const int it  = bid % LT_;
  const int bh  = bid / LT_;
  const int h   = bh % H_;
  const int b0  = bh / H_;

  const float NEG_INF = -__builtin_inff();
  const float scale   = 0.125f;  // 1/sqrt(64), folded into Q

  // ---- -inf fill of score cols beyond the causal tile span (exp -> exact 0) ----
  {
    const int zbase = (it + 1) * 16;
    v8h ninf;
#pragma unroll
    for (int j = 0; j < 8; ++j) ninf[j] = (_Float16)NEG_INF;
    for (int c = zbase + lane * 8; c < S_; c += 32 * 8) {
#pragma unroll
      for (int rr = 0; rr < 16; ++rr)
        *(v8h*)&Sc[rr * SSTR + c] = ninf;
    }
  }

  // ---- Load Q tile (16 x 64) into WMMA A-layout, pre-scaled ----
  const float* qrow = Q + (((size_t)b0 * L_ + (size_t)it * 16 + m) * H_ + h) * E_;
  v16h a0, a1;
  {
    const float* q0 = qrow + 16 * ah;
    const float* q1 = qrow + 16 * ah + 8;
#pragma unroll
    for (int i = 0; i < 8; ++i) {
      a0[i]     = (_Float16)(q0[i]      * scale);
      a0[i + 8] = (_Float16)(q1[i]      * scale);
      a1[i]     = (_Float16)(q0[i + 32] * scale);
      a1[i + 8] = (_Float16)(q1[i + 32] * scale);
    }
  }

  const KT* kbase = K + (((size_t)b0 * S_) * H_ + h) * E_;
  const int row_g = it * 16 + 8 * mh;

  auto qk_tile = [&](int jt) -> v8f {
    const KT* krow = kbase + (size_t)(jt * 16 + n) * (H_ * E_) + 16 * kh;
    v16h kb0 = ld16_cvt(krow);
    v16h kb1 = ld16_cvt(krow + 32);
    v8f c = {};
    c = __builtin_amdgcn_wmma_f32_16x16x32_f16(false, a0, false, kb0, (short)0, c, false, false);
    c = __builtin_amdgcn_wmma_f32_16x16x32_f16(false, a1, false, kb1, (short)0, c, false, false);
    return c;
  };

  // ================= Pass A: QK^T once; scores -> LDS; per-lane running max only =================
  float mrun[8];
#pragma unroll
  for (int r = 0; r < 8; ++r) mrun[r] = NEG_INF;

  // Off-diagonal tiles: no masking needed.
  for (int jt = 0; jt < it; ++jt) {
    __builtin_prefetch(kbase + (size_t)((jt + 1) * 16 + n) * (H_ * E_) + 16 * kh, 0, 1);
    v8f c = qk_tile(jt);
#pragma unroll
    for (int r = 0; r < 8; ++r) {
      float s = c[r];
      mrun[r] = __builtin_elementwise_max(mrun[r], s);
      Sc[(r + 8 * mh) * SSTR + jt * 16 + n] = (_Float16)s;
    }
  }
  // Diagonal tile: apply causal mask.
  {
    v8f c = qk_tile(it);
    const int col_g = it * 16 + n;
#pragma unroll
    for (int r = 0; r < 8; ++r) {
      float s = (col_g > row_g + r) ? NEG_INF : c[r];
      mrun[r] = __builtin_elementwise_max(mrun[r], s);
      Sc[(r + 8 * mh) * SSTR + it * 16 + n] = (_Float16)s;
    }
  }

  // One-time 16-lane max butterfly + broadcast via LDS.
#pragma unroll
  for (int r = 0; r < 8; ++r) {
#pragma unroll
    for (int off = 1; off < 16; off <<= 1)
      mrun[r] = __builtin_elementwise_max(mrun[r], __shfl_xor(mrun[r], off, 32));
    m_lds[r + 8 * mh] = mrun[r];  // group-uniform value, same addr/data per group
  }
  __syncthreads();

  // ================= Pass B1: exp sweep + row sums (unnormalized exp back to LDS) =================
  {
    const int row = lane & 15;
    const int hf  = lane >> 4;
    const float mr = m_lds[row];
    float lsum = 0.0f;
    _Float16* srow = &Sc[row * SSTR + hf * 1024];
    for (int c = 0; c < 1024; c += 8) {
      v8h x = *(v8h*)&srow[c];
      v8h y;
#pragma unroll
      for (int j = 0; j < 8; ++j) {
        float e = __expf((float)x[j] - mr);
        lsum += e;
        y[j] = (_Float16)e;
      }
      *(v8h*)&srow[c] = y;
    }
    lsum += __shfl_xor(lsum, 16, 32);
    r_lds[row] = 1.0f / lsum;  // both half-lanes: same addr, same data
  }
  __syncthreads();

  // ================= Pass B2: O_unnorm = P_unnorm x V (fold 1/sum into O store) =================
  const KT* vbase = V + (((size_t)b0 * S_) * H_ + h) * D_;
  v8f acc[4] = {};
  const int nc = it / 2 + 1;

  for (int cidx = 0; cidx < nc; ++cidx) {
    const int s0 = cidx * 32;

    // Stage V chunk [32 x 64] -> LDS f16, coalesced vector loads.
#pragma unroll
    for (int ig = 0; ig < 4; ++ig) {
      const int sl    = ig * 8 + (lane >> 2);
      const int dbase = (lane & 3) * 16;
      v16h t = ld16_cvt(vbase + (size_t)(s0 + sl) * (H_ * D_) + dbase);
      _Float16* vrow = &Vh[sl * VSTR + dbase];
      *(v4h*)(vrow + 0)  = __builtin_shufflevector(t, t, 0, 1, 2, 3);
      *(v4h*)(vrow + 4)  = __builtin_shufflevector(t, t, 4, 5, 6, 7);
      *(v4h*)(vrow + 8)  = __builtin_shufflevector(t, t, 8, 9, 10, 11);
      *(v4h*)(vrow + 12) = __builtin_shufflevector(t, t, 12, 13, 14, 15);
    }
    __syncthreads();

    // P in A-layout: two b128 LDS loads, zero math.
    const _Float16* prow = &Sc[m * SSTR + s0];
    v8h pl0 = *(const v8h*)&prow[16 * ah];
    v8h pl1 = *(const v8h*)&prow[16 * ah + 8];
    v16h pa = __builtin_shufflevector(pl0, pl1, 0, 1, 2, 3, 4, 5, 6, 7,
                                      8, 9, 10, 11, 12, 13, 14, 15);

#pragma unroll
    for (int dt = 0; dt < 4; ++dt) {
      v16h vb;  // B-layout: element i -> key s0+16*kh+i, column d = 16*dt+n
#pragma unroll
      for (int i = 0; i < 16; ++i)
        vb[i] = Vh[(16 * kh + i) * VSTR + 16 * dt + n];
      acc[dt] = __builtin_amdgcn_wmma_f32_16x16x32_f16(false, pa, false, vb, (short)0,
                                                       acc[dt], false, false);
    }
    __syncthreads();
  }

  // ================= Write O (normalize here) =================
  {
    float rr[8];
#pragma unroll
    for (int r = 0; r < 8; ++r) rr[r] = r_lds[r + 8 * mh];
    float* obase = outO + (((size_t)b0 * L_ + (size_t)it * 16) * H_ + h) * D_;
#pragma unroll
    for (int dt = 0; dt < 4; ++dt)
#pragma unroll
      for (int r = 0; r < 8; ++r)
        obase[(size_t)(r + 8 * mh) * (H_ * D_) + 16 * dt + n] = acc[dt][r] * rr[r];
  }

  // ================= Write A: normalized, coalesced b128 stores =================
  {
    float* Arow = outA + (((size_t)b0 * H_ + h) * L_ + (size_t)it * 16) * S_;
    for (int row = 0; row < 16; ++row) {
      const float ri = r_lds[row];
      const _Float16* srow = &Sc[row * SSTR];
      float* ga = Arow + (size_t)row * S_;
      for (int c = lane * 4; c < S_; c += 128) {
        v4h xh = *(const v4h*)&srow[c];
        v4f xf;
#pragma unroll
        for (int j = 0; j < 4; ++j) xf[j] = (float)xh[j] * ri;
        *(v4f*)(ga + c) = xf;
      }
    }
  }
}

extern "C" void kernel_launch(void* const* d_in, const int* in_sizes, int n_in,
                              void* d_out, int out_size, void* d_ws, size_t ws_size,
                              hipStream_t stream) {
  (void)in_sizes; (void)n_in; (void)out_size;
  const float* Q = (const float*)d_in[0];
  const float* K = (const float*)d_in[1];
  const float* V = (const float*)d_in[2];
  // d_in[3] is the boolean causal mask; the causal structure is computed in-kernel.
  float* outO = (float*)d_out;                       // V_out first
  float* outA = outO + (size_t)B_ * L_ * H_ * D_;    // then A

  const size_t nKV    = (size_t)B_ * S_ * H_ * E_;   // 8,388,608 elements each
  const size_t wsNeed = 2 * nKV * sizeof(_Float16);  // K + V in f16
  dim3 grid(B_ * H_ * LT_);                          // 8192 single-wave workgroups

  if (ws_size >= wsNeed) {
    _Float16* kws = (_Float16*)d_ws;
    _Float16* vws = kws + nKV;
    const int n8 = (int)(nKV / 8);                   // 1,048,576
    cvt_f32_f16<<<(n8 + 255) / 256, 256, 0, stream>>>(K, kws, n8);
    cvt_f32_f16<<<(n8 + 255) / 256, 256, 0, stream>>>(V, vws, n8);
    fa_fwd_causal<_Float16><<<grid, 32, 0, stream>>>(Q, kws, vws, outO, outA);
  } else {
    fa_fwd_causal<float><<<grid, 32, 0, stream>>>(Q, K, V, outO, outA);
  }
}
